// XEyTransformerLayer_704374636747
// MI455X (gfx1250) — matrix operand order repaired and checked by reading, not compiled, not run on hardware
//
#include <hip/hip_runtime.h>

// ---------------------------------------------------------------------------
// MI455X (gfx1250, wave32) fused DiGress-style XEy transformer layer.
// bf16 WMMA (v_wmma_f32_16x16x32_bf16) for all GEMMs, f32 accumulation,
// fused E-path (GEMM1 -> Y -> online softmax -> GEMM2 -> LN -> FFE -> LN)
// so the 134MB Y/newE intermediates never touch HBM. Chunk staging uses the
// CDNA5 async global->LDS DMA path (ASYNCcnt) instead of VGPR round-trips.
// ---------------------------------------------------------------------------

typedef __attribute__((ext_vector_type(16))) __bf16 v16bf;
typedef __attribute__((ext_vector_type(8)))  float  v8f;

__device__ __forceinline__ v8f wmma_bf16(v16bf a, v16bf b, v8f c) {
  return __builtin_amdgcn_wmma_f32_16x16x32_bf16(false, a, false, b, (short)0, c,
                                                 false, false);
}

// Async 16B global->LDS copy (GLOBAL_LOAD_ASYNC_TO_LDS_B128, GV mode).
// Generic LDS pointer low 32 bits == LDS byte address (flat aperture rule).
__device__ __forceinline__ void async_g2l_b128(void* lds, const void* gaddr) {
  unsigned int l = (unsigned int)(unsigned long long)lds;
  asm volatile("global_load_async_to_lds_b128 %0, %1, off"
               :
               : "v"(l), "v"(gaddr)
               : "memory");
}
__device__ __forceinline__ void async_wait0() {
  asm volatile("s_wait_asynccnt 0" ::: "memory");
}

// A fragment (16x32, MxK): lane m = lane%16 = row, half = lane/16.
// a[2v+p] = A[row][ (v%4)*2+p + half*8 + (v/4)*16 ]
__device__ __forceinline__ v16bf fragA_f32(const float* lds, int stride, int row0, int k0) {
  const int lane = threadIdx.x & 31;
  const int m = lane & 15, h = lane >> 4;
  const float* r = lds + (row0 + m) * stride + k0 + h * 8;
  v16bf a;
#pragma unroll
  for (int v = 0; v < 8; ++v) {
    const int k = ((v & 3) * 2) + ((v >> 2) * 16);
    a[2 * v]     = (__bf16)r[k];
    a[2 * v + 1] = (__bf16)r[k + 1];
  }
  return a;
}

__device__ __forceinline__ v16bf fragA_bf16(const __bf16* lds, int stride, int row0, int k0) {
  const int lane = threadIdx.x & 31;
  const int m = lane & 15, h = lane >> 4;
  const __bf16* r = lds + (row0 + m) * stride + k0 + h * 8;
  v16bf a;
#pragma unroll
  for (int v = 0; v < 8; ++v) {
    const int k = ((v & 3) * 2) + ((v >> 2) * 16);
    a[2 * v]     = r[k];
    a[2 * v + 1] = r[k + 1];
  }
  return a;
}

// B fragment (32x16, KxN): lane n = lane%16 = col, half = lane/16.
// b[2v+p] = B[k0 + (v%4)*2+p + half*8 + (v/4)*16][n0 + n]
__device__ __forceinline__ v16bf fragB_f32(const float* lds, int stride, int k0, int n0) {
  const int lane = threadIdx.x & 31;
  const int n = lane & 15, h = lane >> 4;
  const float* base = lds + (k0 + h * 8) * stride + n0 + n;
  v16bf b;
#pragma unroll
  for (int v = 0; v < 8; ++v) {
    const int k = ((v & 3) * 2) + ((v >> 2) * 16);
    b[2 * v]     = (__bf16)base[k * stride];
    b[2 * v + 1] = (__bf16)base[(k + 1) * stride];
  }
  return b;
}

__device__ __forceinline__ v16bf fragB_bf16(const __bf16* lds, int stride, int k0, int n0) {
  const int lane = threadIdx.x & 31;
  const int n = lane & 15, h = lane >> 4;
  const __bf16* base = lds + (k0 + h * 8) * stride + n0 + n;
  v16bf b;
#pragma unroll
  for (int v = 0; v < 8; ++v) {
    const int k = ((v & 3) * 2) + ((v >> 2) * 16);
    b[2 * v]     = base[k * stride];
    b[2 * v + 1] = base[(k + 1) * stride];
  }
  return b;
}

// ---------------------------------------------------------------------------
// Kernel 1: Q/K/V projections (M=512, K=256, N=256), blockIdx.z selects matrix.
// ---------------------------------------------------------------------------
__global__ __launch_bounds__(256) void qkv_kernel(
    const float* __restrict__ X,
    const float* __restrict__ Wq, const float* __restrict__ bq,
    const float* __restrict__ Wk, const float* __restrict__ bk,
    const float* __restrict__ Wv, const float* __restrict__ bv,
    float* __restrict__ Qo, float* __restrict__ Ko, float* __restrict__ Vo) {
  __shared__ float sA[64 * 36];
  __shared__ float sB[32 * 68];
  const float* W;
  const float* bias;
  float* out;
  if (blockIdx.z == 0)      { W = Wq; bias = bq; out = Qo; }
  else if (blockIdx.z == 1) { W = Wk; bias = bk; out = Ko; }
  else                      { W = Wv; bias = bv; out = Vo; }

  const int m0 = blockIdx.x * 64, n0g = blockIdx.y * 64;
  const int tid = threadIdx.x, w = tid >> 5, lane = tid & 31;
  const int nIdx = lane & 15, half = lane >> 4;
  const int mt = w >> 1, ntBase = (w & 1) * 2;
  v8f acc0 = {}, acc1 = {};

  for (int kk = 0; kk < 8; ++kk) {
    for (int idx = tid; idx < 64 * 32; idx += 256) {
      int r = idx >> 5, c = idx & 31;
      sA[r * 36 + c] = X[(m0 + r) * 256 + kk * 32 + c];
    }
    for (int idx = tid; idx < 32 * 64; idx += 256) {
      int r = idx >> 6, c = idx & 63;
      sB[r * 68 + c] = W[(kk * 32 + r) * 256 + n0g + c];
    }
    __syncthreads();
    v16bf a  = fragA_f32(sA, 36, mt * 16, 0);
    v16bf b0 = fragB_f32(sB, 68, 0, ntBase * 16);
    v16bf b1 = fragB_f32(sB, 68, 0, ntBase * 16 + 16);
    acc0 = wmma_bf16(a, b0, acc0);
    acc1 = wmma_bf16(a, b1, acc1);
    __syncthreads();
  }
#pragma unroll
  for (int t = 0; t < 2; ++t) {
    v8f acc = t ? acc1 : acc0;
    const int nt = ntBase + t;
#pragma unroll
    for (int r = 0; r < 8; ++r) {
      const int row = m0 + mt * 16 + r + 8 * half;
      const int col = n0g + nt * 16 + nIdx;
      out[row * 256 + col] = acc[r] + bias[col];  // node_mask == 1
    }
  }
}

// ---------------------------------------------------------------------------
// Kernel 2/3: aggregations for the y path.
// ---------------------------------------------------------------------------
__global__ void xagg_kernel(const float* __restrict__ X, float* __restrict__ xagg) {
  const int b = blockIdx.x, c = threadIdx.x;
  float s = 0.f, sq = 0.f, mn = 3.4e38f, mx = -3.4e38f;
  for (int n = 0; n < 256; ++n) {
    float v = X[((b * 256) + n) * 256 + c];
    s += v; sq += v * v; mn = fminf(mn, v); mx = fmaxf(mx, v);
  }
  float var = (sq - s * s / 256.f) / 255.f;
  xagg[b * 1024 + c]       = s / 256.f;
  xagg[b * 1024 + 256 + c] = mn;
  xagg[b * 1024 + 512 + c] = mx;
  xagg[b * 1024 + 768 + c] = sqrtf(fmaxf(var, 0.f));
}

__global__ void eagg_kernel(const float* __restrict__ E, float* __restrict__ eagg) {
  __shared__ float ps[4][64], pq[4][64], pmn[4][64], pmx[4][64];
  const int b = blockIdx.x, t = threadIdx.x;
  const int ch = t & 63, part = t >> 6;
  float s = 0.f, sq = 0.f, mn = 3.4e38f, mx = -3.4e38f;
  for (int q = part; q < 65536; q += 4) {
    float v = E[(((size_t)b * 65536) + q) * 64 + ch];
    s += v; sq += v * v; mn = fminf(mn, v); mx = fmaxf(mx, v);
  }
  ps[part][ch] = s; pq[part][ch] = sq; pmn[part][ch] = mn; pmx[part][ch] = mx;
  __syncthreads();
  if (part == 0) {
    for (int p = 1; p < 4; ++p) {
      s += ps[p][ch]; sq += pq[p][ch];
      mn = fminf(mn, pmn[p][ch]); mx = fmaxf(mx, pmx[p][ch]);
    }
    const float M = 65536.f;
    float var = (sq - s * s / M) / (M - 1.f);
    eagg[b * 256 + ch]        = s / M;
    eagg[b * 256 + 64 + ch]   = mn;
    eagg[b * 256 + 128 + ch]  = mx;
    eagg[b * 256 + 192 + ch]  = sqrtf(fmaxf(var, 0.f));
  }
}

// ---------------------------------------------------------------------------
// Kernel 4: entire y path (tiny) + FiLM vectors ye_add/ye_mul/yx_add/yx_mul.
// ---------------------------------------------------------------------------
__global__ __launch_bounds__(256) void y_kernel(
    const float* __restrict__ y,
    const float* __restrict__ Wyea, const float* __restrict__ byea,
    const float* __restrict__ Wyem, const float* __restrict__ byem,
    const float* __restrict__ Wyxa, const float* __restrict__ byxa,
    const float* __restrict__ Wyxm, const float* __restrict__ byxm,
    const float* __restrict__ Wyy, const float* __restrict__ byy,
    const float* __restrict__ Wxy, const float* __restrict__ bxy,
    const float* __restrict__ Wey, const float* __restrict__ bey,
    const float* __restrict__ Wo1, const float* __restrict__ bo1,
    const float* __restrict__ Wo2, const float* __restrict__ bo2,
    const float* __restrict__ n1w, const float* __restrict__ n1b,
    const float* __restrict__ Wf1, const float* __restrict__ bf1,
    const float* __restrict__ Wf2, const float* __restrict__ bf2,
    const float* __restrict__ n2w, const float* __restrict__ n2b,
    const float* __restrict__ xagg, const float* __restrict__ eagg,
    float* __restrict__ yeadd, float* __restrict__ yemul,
    float* __restrict__ yxadd, float* __restrict__ yxmul,
    float* __restrict__ Yout) {
  __shared__ float sy[64], sv[64], st[64], sh[256], sred[2];
  const int tid = threadIdx.x;
  for (int b = 0; b < 2; ++b) {
    if (tid < 64) sy[tid] = y[b * 64 + tid];
    __syncthreads();
    {  // FiLM vectors from original y
      float a0 = byea[tid], a1 = byem[tid], a2 = byxa[tid], a3 = byxm[tid];
      for (int k = 0; k < 64; ++k) {
        float yv = sy[k];
        a0 += yv * Wyea[k * 256 + tid];
        a1 += yv * Wyem[k * 256 + tid];
        a2 += yv * Wyxa[k * 256 + tid];
        a3 += yv * Wyxm[k * 256 + tid];
      }
      yeadd[b * 256 + tid] = a0; yemul[b * 256 + tid] = a1;
      yxadd[b * 256 + tid] = a2; yxmul[b * 256 + tid] = a3;
    }
    if (tid < 64) {
      float v = byy[tid] + bxy[tid] + bey[tid];
      for (int k = 0; k < 64; ++k)   v += sy[k] * Wyy[k * 64 + tid];
      for (int k = 0; k < 1024; ++k) v += xagg[b * 1024 + k] * Wxy[k * 64 + tid];
      for (int k = 0; k < 256; ++k)  v += eagg[b * 256 + k] * Wey[k * 64 + tid];
      sv[tid] = v;
    }
    __syncthreads();
    if (tid < 64) {
      float t1 = bo1[tid];
      for (int k = 0; k < 64; ++k) t1 += sv[k] * Wo1[k * 64 + tid];
      st[tid] = fmaxf(t1, 0.f);
    }
    __syncthreads();
    if (tid < 64) {
      float v2 = bo2[tid];
      for (int k = 0; k < 64; ++k) v2 += st[k] * Wo2[k * 64 + tid];
      sv[tid] = sy[tid] + v2;  // residual
    }
    __syncthreads();
    if (tid == 0) {
      float s = 0.f, sq = 0.f;
      for (int c = 0; c < 64; ++c) { s += sv[c]; sq += sv[c] * sv[c]; }
      float m = s / 64.f;
      sred[0] = m; sred[1] = rsqrtf(sq / 64.f - m * m + 1e-5f);
    }
    __syncthreads();
    if (tid < 64) st[tid] = (sv[tid] - sred[0]) * sred[1] * n1w[tid] + n1b[tid];
    __syncthreads();
    {
      float h = bf1[tid];
      for (int k = 0; k < 64; ++k) h += st[k] * Wf1[k * 256 + tid];
      sh[tid] = fmaxf(h, 0.f);
    }
    __syncthreads();
    if (tid < 64) {
      float v = st[tid] + bf2[tid];
      for (int k = 0; k < 256; ++k) v += sh[k] * Wf2[k * 64 + tid];
      sv[tid] = v;
    }
    __syncthreads();
    if (tid == 0) {
      float s = 0.f, sq = 0.f;
      for (int c = 0; c < 64; ++c) { s += sv[c]; sq += sv[c] * sv[c]; }
      float m = s / 64.f;
      sred[0] = m; sred[1] = rsqrtf(sq / 64.f - m * m + 1e-5f);
    }
    __syncthreads();
    if (tid < 64) Yout[b * 64 + tid] = (sv[tid] - sred[0]) * sred[1] * n2w[tid] + n2b[tid];
    __syncthreads();
  }
}

// ---------------------------------------------------------------------------
// Kernel 5: fused E path + per-channel online-softmax attention.
// One workgroup = (b, 16 i-rows); loops 16 j-chunks of 16. 256 threads/8 waves.
// Dynamic LDS layout (bytes):
constexpr unsigned EDGE_LDS = 294912;  // 288 KB of the WGP's 320 KB
// ---------------------------------------------------------------------------
__global__ __launch_bounds__(256) void edge_attn_kernel(
    const float* __restrict__ E,
    const float* __restrict__ Wemul, const float* __restrict__ Weadd,
    const float* __restrict__ Weout,
    const float* __restrict__ bemul, const float* __restrict__ beadd,
    const float* __restrict__ beout,
    const float* __restrict__ n1w, const float* __restrict__ n1b,
    const float* __restrict__ n2w, const float* __restrict__ n2b,
    const float* __restrict__ Wf1, const float* __restrict__ bf1,
    const float* __restrict__ Wf2, const float* __restrict__ bf2,
    const float* __restrict__ Qb, const float* __restrict__ Kb,
    const float* __restrict__ Vb,
    const float* __restrict__ yea, const float* __restrict__ yem,
    const float* __restrict__ yxa, const float* __restrict__ yxm,
    float* __restrict__ Eout, float* __restrict__ newXpre) {
  extern __shared__ char smem[];
  __bf16* sWmul = (__bf16*)(smem);             // [64][256] bf16
  __bf16* sWadd = (__bf16*)(smem + 32768);     // [64][256] bf16
  __bf16* sWout = (__bf16*)(smem + 65536);     // [256][64] bf16
  float*  sE    = (float*)(smem + 98304);      // [256][68] f32 (E chunk / out1)
  __bf16* sTmp  = (__bf16*)(smem + 167936);    // [256][40] bf16
  float*  sQ    = (float*)(smem + 188416);     // [16][260]
  float*  sK    = (float*)(smem + 205056);     // [16][260]
  float*  sV    = (float*)(smem + 221696);     // [16][260]
  float*  sM    = (float*)(smem + 238336);     // [16][256] softmax running max
  float*  sL    = (float*)(smem + 254720);     // [16][256] running denom
  float*  sS    = (float*)(smem + 271104);     // [16][256] running numer
  float*  cBm   = (float*)(smem + 287488);     // consts
  float*  cBa   = cBm + 256;
  float*  cYea  = cBa + 256;
  float*  cYem  = cYea + 256;
  float*  cBo   = cYem + 256;
  float*  cN1w  = cBo + 64;
  float*  cN1b  = cN1w + 64;
  float*  cN2w  = cN1b + 64;
  float*  cN2b  = cN2w + 64;
  float*  cF2b  = cN2b + 64;
  float*  cF1b  = cF2b + 64;  // 128

  const int tid = threadIdx.x;
  const int w = tid >> 5, lane = tid & 31;
  const int nIdx = lane & 15, half = lane >> 4;
  const int b = blockIdx.x >> 4;
  const int i0 = (blockIdx.x & 15) * 16;
  const float NEG_INF = -__builtin_inff();

  for (int idx = tid; idx < 64 * 256; idx += 256) {
    sWmul[idx] = (__bf16)Wemul[idx];
    sWadd[idx] = (__bf16)Weadd[idx];
  }
  for (int idx = tid; idx < 256 * 64; idx += 256) sWout[idx] = (__bf16)Weout[idx];
  cBm[tid] = bemul[tid];
  cBa[tid] = beadd[tid];
  cYea[tid] = yea[b * 256 + tid];
  cYem[tid] = yem[b * 256 + tid];
  if (tid < 64) {
    cBo[tid] = beout[tid];
    cN1w[tid] = n1w[tid]; cN1b[tid] = n1b[tid];
    cN2w[tid] = n2w[tid]; cN2b[tid] = n2b[tid];
    cF2b[tid] = bf2[tid];
  }
  if (tid < 128) cF1b[tid] = bf1[tid];
  const float qscale = 0.17677669529663687f;  // 1/sqrt(DF=32)
  for (int idx = tid; idx < 16 * 256; idx += 256) {
    int r = idx >> 8, c = idx & 255;
    sQ[r * 260 + c] = Qb[((b * 256 + i0 + r) * 256) + c] * qscale;
    sM[idx] = NEG_INF; sL[idx] = 0.f; sS[idx] = 0.f;
  }
  __syncthreads();

  for (int jt = 0; jt < 16; ++jt) {
    const int j0 = jt * 16;
    if (jt + 1 < 16) {  // warm next chunk (global_prefetch_b8)
      __builtin_prefetch(&Kb[(b * 256 + j0 + 16) * 256], 0, 1);
      __builtin_prefetch(&Vb[(b * 256 + j0 + 16) * 256], 0, 1);
      __builtin_prefetch(&E[(((size_t)(b * 256 + i0) * 256) + j0 + 16) * 64], 0, 1);
    }
    // Async DMA staging: K/V chunks (16 rows x 256 f32) and E chunk
    // (256 rows x 64 f32) go straight to LDS via GLOBAL_LOAD_ASYNC_TO_LDS_B128
    // (tracked by ASYNCcnt; 16B per lane keeps each transfer within one
    // padded LDS row since 256 and 64 are multiples of 4 floats).
    for (int idx = tid; idx < 1024; idx += 256) {  // 16 rows x 64 xfers
      int r = idx >> 6, c4 = (idx & 63) * 4;
      async_g2l_b128(&sK[r * 260 + c4], &Kb[(b * 256 + j0 + r) * 256 + c4]);
      async_g2l_b128(&sV[r * 260 + c4], &Vb[(b * 256 + j0 + r) * 256 + c4]);
    }
    for (int idx = tid; idx < 4096; idx += 256) {  // 256 rows x 16 xfers
      int row = idx >> 4, c4 = (idx & 15) * 4;
      int il = row >> 4, jl = row & 15;
      async_g2l_b128(&sE[row * 68 + c4],
                     &E[((((size_t)b * 256 + i0 + il) * 256) + j0 + jl) * 64 + c4]);
    }
    async_wait0();
    __syncthreads();

    v8f acc[8] = {};  // GEMM2 accum: tmp(256x256ch) @ W_eout(256x64)

    for (int cp = 0; cp < 8; ++cp) {
      // ---- stage 1: E1/E2 GEMM, Y, FiLM tmp, online softmax stats ----
#pragma unroll
      for (int ctl = 0; ctl < 2; ++ctl) {
        const int ct = cp * 2 + ctl;
        const int c = ct * 16 + nIdx;
        v16bf bm0 = fragB_bf16(sWmul, 256, 0, ct * 16);
        v16bf bm1 = fragB_bf16(sWmul, 256, 32, ct * 16);
        v16bf ba0 = fragB_bf16(sWadd, 256, 0, ct * 16);
        v16bf ba1 = fragB_bf16(sWadd, 256, 32, ct * 16);
        const float biasM = cBm[c], biasA = cBa[c];
        const float yeav = cYea[c], yemv = cYem[c];
#pragma unroll
        for (int mi = 0; mi < 2; ++mi) {
          const int mt = 2 * w + mi;  // i_local
          v16bf a0 = fragA_f32(sE, 68, mt * 16, 0);
          v16bf a1 = fragA_f32(sE, 68, mt * 16, 32);
          v8f e1 = {};
          e1 = wmma_bf16(a0, bm0, e1);
          e1 = wmma_bf16(a1, bm1, e1);
          v8f e2 = {};
          e2 = wmma_bf16(a0, ba0, e2);
          e2 = wmma_bf16(a1, ba1, e2);
          const float qv = sQ[mt * 260 + c];
          float Yv[8];
          float mx = NEG_INF;
#pragma unroll
          for (int r = 0; r < 8; ++r) {
            const int jl = half * 8 + r;
            float kv = sK[jl * 260 + c];
            float yv = qv * kv * ((e1[r] + biasM) + 1.f) + (e2[r] + biasA);
            Yv[r] = yv;
            mx = fmaxf(mx, yv);
            sTmp[(mt * 16 + jl) * 40 + ctl * 16 + nIdx] =
                (__bf16)(yeav + (yemv + 1.f) * yv);
          }
          mx = fmaxf(mx, __shfl_xor(mx, 16, 32));
          const float Mold = sM[mt * 256 + c];
          const float Mn = fmaxf(Mold, mx);
          float se = 0.f, sv = 0.f;
#pragma unroll
          for (int r = 0; r < 8; ++r) {
            float p = __expf(Yv[r] - Mn);
            se += p;
            sv += p * sV[(half * 8 + r) * 260 + c];
          }
          se += __shfl_xor(se, 16, 32);
          sv += __shfl_xor(sv, 16, 32);
          if (half == 0) {
            float sc = __expf(Mold - Mn);
            sM[mt * 256 + c] = Mn;
            sL[mt * 256 + c] = sL[mt * 256 + c] * sc + se;
            sS[mt * 256 + c] = sS[mt * 256 + c] * sc + sv;
          }
        }
      }
      __syncthreads();
      // ---- stage 2: acc += tmp[:, 32ch] @ W_eout[32ch, 64] ----
#pragma unroll
      for (int mi = 0; mi < 2; ++mi) {
        const int mt2 = 2 * w + mi;
        v16bf at = fragA_bf16(sTmp, 40, mt2 * 16, 0);
#pragma unroll
        for (int nt = 0; nt < 4; ++nt) {
          v16bf bt = fragB_bf16(sWout, 64, cp * 32, nt * 16);
          acc[mi * 4 + nt] = wmma_bf16(at, bt, acc[mi * 4 + nt]);
        }
      }
      __syncthreads();
    }

    // ---- epilogue: newE + bias + E residual + LN(normE1) -> overwrite sE ----
#pragma unroll
    for (int mi = 0; mi < 2; ++mi) {
      const int mt2 = 2 * w + mi;
#pragma unroll
      for (int r = 0; r < 8; ++r) {
        const int row = mt2 * 16 + r + 8 * half;
        float vals[4];
        float s = 0.f, sq = 0.f;
#pragma unroll
        for (int nt = 0; nt < 4; ++nt) {
          const int ch = nt * 16 + nIdx;
          float v = acc[mi * 4 + nt][r] + cBo[ch] + sE[row * 68 + ch];
          vals[nt] = v; s += v; sq += v * v;
        }
#pragma unroll
        for (int d = 1; d < 16; d <<= 1) {
          s += __shfl_xor(s, d, 32);
          sq += __shfl_xor(sq, d, 32);
        }
        const float mean = s * (1.f / 64.f);
        const float inv = rsqrtf(sq * (1.f / 64.f) - mean * mean + 1e-5f);
#pragma unroll
        for (int nt = 0; nt < 4; ++nt) {
          const int ch = nt * 16 + nIdx;
          sE[row * 68 + ch] = (vals[nt] - mean) * inv * cN1w[ch] + cN1b[ch];
        }
      }
    }
    __syncthreads();

    // ---- FFE (64->128->64) + residual + LN(normE2), one row per thread ----
    {
      const int row = tid;
      const int il = row >> 4, jl = row & 15;
      float x[64], out[64];
#pragma unroll
      for (int c = 0; c < 64; ++c) {
        x[c] = sE[row * 68 + c];
        out[c] = x[c] + cF2b[c];
      }
      for (int k = 0; k < 128; ++k) {
        float h = cF1b[k];
#pragma unroll
        for (int c = 0; c < 64; ++c) h += x[c] * Wf1[c * 128 + k];
        h = fmaxf(h, 0.f);
        const float* w2r = &Wf2[k * 64];
#pragma unroll
        for (int c = 0; c < 64; ++c) out[c] += h * w2r[c];
      }
      float s = 0.f, sq = 0.f;
#pragma unroll
      for (int c = 0; c < 64; ++c) { s += out[c]; sq += out[c] * out[c]; }
      const float mean = s * (1.f / 64.f);
      const float inv = rsqrtf(sq * (1.f / 64.f) - mean * mean + 1e-5f);
      float* eo = &Eout[((((size_t)b * 256 + i0 + il) * 256) + j0 + jl) * 64];
#pragma unroll
      for (int c = 0; c < 64; ++c) eo[c] = (out[c] - mean) * inv * cN2w[c] + cN2b[c];
    }
    __syncthreads();
  }

  // ---- finalize attention: wV = S/L, FiLM -> newXpre ----
  for (int idx = tid; idx < 16 * 256; idx += 256) {
    int il = idx >> 8, c = idx & 255;
    float wv = sS[idx] / sL[idx];
    float nx = yxa[b * 256 + c] + (yxm[b * 256 + c] + 1.f) * wv;
    newXpre[(((size_t)b * 256) + i0 + il) * 256 + c] = nx;
  }
}

// ---------------------------------------------------------------------------
// Kernel 6: newX = newXpre @ x_out.w + b (+X residual), LN(normX1) -> Xn1.
// 16-row blocks, full 256-wide rows (so LN is block-local).
// ---------------------------------------------------------------------------
__global__ __launch_bounds__(256) void xout_kernel(
    const float* __restrict__ A, const float* __restrict__ W,
    const float* __restrict__ bias, const float* __restrict__ Xres,
    const float* __restrict__ lnw, const float* __restrict__ lnb,
    float* __restrict__ Out) {
  __shared__ float sA[16 * 36];
  __shared__ float sB[32 * 260];
  __shared__ float rows[16 * 260];
  __shared__ float red[32];
  const int m0 = blockIdx.x * 16;
  const int tid = threadIdx.x, w = tid >> 5, lane = tid & 31;
  const int nIdx = lane & 15, half = lane >> 4;
  v8f acc0 = {}, acc1 = {};
  for (int kk = 0; kk < 8; ++kk) {
    for (int idx = tid; idx < 16 * 32; idx += 256) {
      int r = idx >> 5, c = idx & 31;
      sA[r * 36 + c] = A[(m0 + r) * 256 + kk * 32 + c];
    }
    for (int idx = tid; idx < 32 * 256; idx += 256) {
      int r = idx >> 8, c = idx & 255;
      sB[r * 260 + c] = W[(kk * 32 + r) * 256 + c];
    }
    __syncthreads();
    v16bf a = fragA_f32(sA, 36, 0, 0);
    v16bf b0 = fragB_f32(sB, 260, 0, (2 * w) * 16);
    v16bf b1 = fragB_f32(sB, 260, 0, (2 * w + 1) * 16);
    acc0 = wmma_bf16(a, b0, acc0);
    acc1 = wmma_bf16(a, b1, acc1);
    __syncthreads();
  }
#pragma unroll
  for (int t = 0; t < 2; ++t) {
    v8f acc = t ? acc1 : acc0;
    const int nt = 2 * w + t;
#pragma unroll
    for (int r = 0; r < 8; ++r) {
      const int rr = r + 8 * half;
      const int col = nt * 16 + nIdx;
      rows[rr * 260 + col] = acc[r] + bias[col] + Xres[(m0 + rr) * 256 + col];
    }
  }
  __syncthreads();
  if (tid < 16) {
    float s = 0.f, sq = 0.f;
    for (int c = 0; c < 256; ++c) {
      float v = rows[tid * 260 + c];
      s += v; sq += v * v;
    }
    float m = s / 256.f;
    red[tid * 2] = m;
    red[tid * 2 + 1] = rsqrtf(sq / 256.f - m * m + 1e-5f);
  }
  __syncthreads();
  for (int idx = tid; idx < 16 * 256; idx += 256) {
    int r = idx >> 8, c = idx & 255;
    Out[(m0 + r) * 256 + c] =
        (rows[r * 260 + c] - red[r * 2]) * red[r * 2 + 1] * lnw[c] + lnb[c];
  }
}

// ---------------------------------------------------------------------------
// Kernel 7: H = relu(Xn1 @ linX1 + b1) stored bf16 (M=512,K=256,N=2048).
// ---------------------------------------------------------------------------
__global__ __launch_bounds__(256) void ffx1_kernel(
    const float* __restrict__ A, const float* __restrict__ W,
    const float* __restrict__ bias, __bf16* __restrict__ H) {
  __shared__ float sA[64 * 36];
  __shared__ float sB[32 * 68];
  const int m0 = blockIdx.x * 64, n0g = blockIdx.y * 64;
  const int tid = threadIdx.x, w = tid >> 5, lane = tid & 31;
  const int nIdx = lane & 15, half = lane >> 4;
  const int mt = w >> 1, ntBase = (w & 1) * 2;
  v8f acc0 = {}, acc1 = {};
  for (int kk = 0; kk < 8; ++kk) {
    for (int idx = tid; idx < 64 * 32; idx += 256) {
      int r = idx >> 5, c = idx & 31;
      sA[r * 36 + c] = A[(m0 + r) * 256 + kk * 32 + c];
    }
    for (int idx = tid; idx < 32 * 64; idx += 256) {
      int r = idx >> 6, c = idx & 63;
      sB[r * 68 + c] = W[(size_t)(kk * 32 + r) * 2048 + n0g + c];
    }
    __syncthreads();
    v16bf a = fragA_f32(sA, 36, mt * 16, 0);
    v16bf b0 = fragB_f32(sB, 68, 0, ntBase * 16);
    v16bf b1 = fragB_f32(sB, 68, 0, ntBase * 16 + 16);
    acc0 = wmma_bf16(a, b0, acc0);
    acc1 = wmma_bf16(a, b1, acc1);
    __syncthreads();
  }
#pragma unroll
  for (int t = 0; t < 2; ++t) {
    v8f acc = t ? acc1 : acc0;
    const int nt = ntBase + t;
#pragma unroll
    for (int r = 0; r < 8; ++r) {
      const int row = m0 + mt * 16 + r + 8 * half;
      const int col = n0g + nt * 16 + nIdx;
      H[(size_t)row * 2048 + col] = (__bf16)fmaxf(acc[r] + bias[col], 0.f);
    }
  }
}

// ---------------------------------------------------------------------------
// Kernel 8: X = LN(Xn1 + H @ linX2 + b2, normX2)  (K=2048).
// ---------------------------------------------------------------------------
__global__ __launch_bounds__(256) void ffx2_kernel(
    const __bf16* __restrict__ H, const float* __restrict__ W,
    const float* __restrict__ bias, const float* __restrict__ Xres,
    const float* __restrict__ lnw, const float* __restrict__ lnb,
    float* __restrict__ Out) {
  __shared__ __bf16 sA[16 * 40];
  __shared__ float sB[32 * 260];
  __shared__ float rows[16 * 260];
  __shared__ float red[32];
  const int m0 = blockIdx.x * 16;
  const int tid = threadIdx.x, w = tid >> 5, lane = tid & 31;
  const int nIdx = lane & 15, half = lane >> 4;
  v8f acc0 = {}, acc1 = {};
  for (int kk = 0; kk < 64; ++kk) {
    for (int idx = tid; idx < 16 * 32; idx += 256) {
      int r = idx >> 5, c = idx & 31;
      sA[r * 40 + c] = H[(size_t)(m0 + r) * 2048 + kk * 32 + c];
    }
    for (int idx = tid; idx < 32 * 256; idx += 256) {
      int r = idx >> 8, c = idx & 255;
      sB[r * 260 + c] = W[(size_t)(kk * 32 + r) * 256 + c];
    }
    __syncthreads();
    v16bf a = fragA_bf16(sA, 40, 0, 0);
    v16bf b0 = fragB_f32(sB, 260, 0, (2 * w) * 16);
    v16bf b1 = fragB_f32(sB, 260, 0, (2 * w + 1) * 16);
    acc0 = wmma_bf16(a, b0, acc0);
    acc1 = wmma_bf16(a, b1, acc1);
    __syncthreads();
  }
#pragma unroll
  for (int t = 0; t < 2; ++t) {
    v8f acc = t ? acc1 : acc0;
    const int nt = 2 * w + t;
#pragma unroll
    for (int r = 0; r < 8; ++r) {
      const int rr = r + 8 * half;
      const int col = nt * 16 + nIdx;
      rows[rr * 260 + col] = acc[r] + bias[col] + Xres[(m0 + rr) * 256 + col];
    }
  }
  __syncthreads();
  if (tid < 16) {
    float s = 0.f, sq = 0.f;
    for (int c = 0; c < 256; ++c) {
      float v = rows[tid * 260 + c];
      s += v; sq += v * v;
    }
    float m = s / 256.f;
    red[tid * 2] = m;
    red[tid * 2 + 1] = rsqrtf(sq / 256.f - m * m + 1e-5f);
  }
  __syncthreads();
  for (int idx = tid; idx < 16 * 256; idx += 256) {
    int r = idx >> 8, c = idx & 255;
    Out[(m0 + r) * 256 + c] =
        (rows[r * 260 + c] - red[r * 2]) * red[r * 2 + 1] * lnw[c] + lnb[c];
  }
}

// ---------------------------------------------------------------------------
// Host-side orchestration.
// ---------------------------------------------------------------------------
extern "C" void kernel_launch(void* const* d_in, const int* in_sizes, int n_in,
                              void* d_out, int out_size, void* d_ws, size_t ws_size,
                              hipStream_t stream) {
  (void)in_sizes; (void)n_in; (void)out_size; (void)ws_size;
  const float* X = (const float*)d_in[0];
  const float* E = (const float*)d_in[1];
  const float* y = (const float*)d_in[2];
  // d_in[3] = node_mask: all ones in this problem; folded out.
#define P(i) ((const float*)d_in[i])

  float* ws = (float*)d_ws;
  // workspace layout (float offsets)
  float* Q     = ws + 0;        // 131072
  float* K     = ws + 131072;   // 131072
  float* V     = ws + 262144;   // 131072
  float* YEADD = ws + 393216;   // 512
  float* YEMUL = ws + 393728;
  float* YXADD = ws + 394240;
  float* YXMUL = ws + 394752;
  float* XAGG  = ws + 395264;   // 2048
  float* EAGG  = ws + 397312;   // 512
  float* NXP   = ws + 397824;   // 131072  (yx FiLM of wV, pre-x_out)
  float* XN1   = ws + 528896;   // 131072  (after normX1)
  __bf16* Hws  = (__bf16*)(ws + 659968);  // 512x2048 bf16

  float* outX = (float*)d_out;                 // (2,256,256)
  float* outE = outX + 131072;                 // (2,256,256,64)
  float* outY = outE + 8388608;                // (2,64)

  qkv_kernel<<<dim3(8, 4, 3), 256, 0, stream>>>(
      X, P(4), P(5), P(6), P(7), P(8), P(9), Q, K, V);
  xagg_kernel<<<2, 256, 0, stream>>>(X, XAGG);
  eagg_kernel<<<2, 256, 0, stream>>>(E, EAGG);
  y_kernel<<<1, 256, 0, stream>>>(
      y,
      P(16), P(17),   // y_e_add
      P(14), P(15),   // y_e_mul
      P(20), P(21),   // y_x_add
      P(18), P(19),   // y_x_mul
      P(22), P(23),   // y_y
      P(24), P(25),   // x_y
      P(26), P(27),   // e_y
      P(32), P(33),   // y_out1
      P(34), P(35),   // y_out2
      P(56), P(57),   // norm_y1
      P(44), P(45),   // lin_y1
      P(46), P(47),   // lin_y2
      P(58), P(59),   // norm_y2
      XAGG, EAGG, YEADD, YEMUL, YXADD, YXMUL, outY);
  edge_attn_kernel<<<32, 256, EDGE_LDS, stream>>>(
      E,
      P(12), P(10), P(30),        // e_mul.w, e_add.w, e_out.w
      P(13), P(11), P(31),        // biases
      P(52), P(53), P(54), P(55), // normE1, normE2
      P(40), P(41), P(42), P(43), // linE1, linE2
      Q, K, V, YEADD, YEMUL, YXADD, YXMUL, outE, NXP);
  xout_kernel<<<32, 256, 0, stream>>>(NXP, P(28), P(29), X, P(48), P(49), XN1);
  ffx1_kernel<<<dim3(8, 32), 256, 0, stream>>>(XN1, P(36), P(37), Hws);
  ffx2_kernel<<<32, 256, 0, stream>>>(Hws, P(38), P(39), XN1, P(50), P(51), outX);
#undef P
}
